// Encoder_60361470378445
// MI455X (gfx1250) — matrix operand order repaired and checked by reading, not compiled, not run on hardware
//
#include <hip/hip_runtime.h>
#include <hip/hip_fp16.h>
#include <math.h>

// Problem constants (match reference)
#define H      2048
#define NSEQ   4096
#define TSTEPS (NSEQ + 2)   // 4098 scan steps
// Persistent-kernel geometry
#define G      128          // one workgroup per WGP; each owns 16 h-elements
#define BLOCK  128          // 4 waves (one per SIMD32); BLOCK == G for parallel poll
#define JPW    16           // h elements per workgroup
#define ROWS   64           // gate rows per workgroup (4 gates x 16)
#define WROW   2056         // padded LDS row stride in fp16 elems (2048 + 8)
#define FLAG_STRIDE 32      // u32s between per-WG flags (128 B: own cache line)

typedef __attribute__((ext_vector_type(16))) _Float16 v16h;
typedef __attribute__((ext_vector_type(8)))  _Float16 v8h;
typedef __attribute__((ext_vector_type(8)))  float    v8f;

__device__ __forceinline__ float sigmoidf_(float x) {
    return 1.0f / (1.0f + __expf(-x));
}

// Zero the double-buffered h and the per-WG barrier flags each launch
// (workspace is poisoned; graph replays must start from a clean state).
__global__ void lstm_init(float* h_glob, unsigned* flags) {
    const int i = blockIdx.x * blockDim.x + threadIdx.x;
    if (i < 2 * H) h_glob[i] = 0.0f;
    if (i < G * FLAG_STRIDE) flags[i] = 0u;
}

__global__ void __launch_bounds__(BLOCK, 1)
lstm_persistent(const float* __restrict__ seq,
                const float* __restrict__ W_ih,
                const float* __restrict__ W_hh,
                const float* __restrict__ b_ih,
                const float* __restrict__ b_hh,
                float* __restrict__ out,       // [H final h][ (TSTEPS-1) x H hs ]
                float* __restrict__ h_glob,    // 2*H double buffer
                unsigned* __restrict__ flags)  // G padded arrival flags
{
    extern __shared__ char smem[];
    _Float16* lds_w    = (_Float16*)smem;                                  // ROWS*WROW fp16
    _Float16* lds_h    = (_Float16*)(smem + (size_t)ROWS * WROW * 2);      // H fp16
    float*    lds_gate = (float*)(smem + (size_t)ROWS * WROW * 2 + H * 2); // 64
    float*    lds_c    = lds_gate + 64;                                    // 16
    float*    lds_bias = lds_c + 16;                                       // 64
    float*    lds_wih  = lds_bias + 64;                                    // 192

    const int wg   = blockIdx.x;
    const int tid  = threadIdx.x;
    const int wave = tid >> 5;     // wave 0..3 -> gates i,f,g,o
    const int lane = tid & 31;

    // ---- one-time: stage this WGP's 64 gate rows of W_hh into LDS (fp32 -> fp16)
    for (int idx = tid; idx < ROWS * (H / 4); idx += BLOCK) {
        const int lr = idx >> 9;              // / 512 float4 per row
        const int c  = (idx & 511) << 2;
        const int gr = ((lr >> 4) * H) + wg * JPW + (lr & 15);
        const float4 w = *(const float4*)(W_hh + (size_t)gr * H + c);
        _Float16* dst = lds_w + (size_t)lr * WROW + c;
        dst[0] = (_Float16)w.x; dst[1] = (_Float16)w.y;
        dst[2] = (_Float16)w.z; dst[3] = (_Float16)w.w;
    }
    for (int lr = tid; lr < ROWS; lr += BLOCK) {
        const int gr = ((lr >> 4) * H) + wg * JPW + (lr & 15);
        lds_bias[lr]      = b_ih[gr] + b_hh[gr];
        lds_wih[lr*3 + 0] = W_ih[gr*3 + 0];
        lds_wih[lr*3 + 1] = W_ih[gr*3 + 1];
        lds_wih[lr*3 + 2] = W_ih[gr*3 + 2];
    }
    if (tid < JPW) lds_c[tid] = 0.0f;
    __syncthreads();

    // WMMA A-operand addressing (16x32 fp16 A, wave32; contiguous k order chosen here, B matches):
    //   lanes 0-15 : row = lane,    K = {0..15}
    //   lanes 16-31: row = lane-16, K = {16..31}
    const int row = lane & 15;
    const int hi  = lane >> 4;
    const _Float16* wrow = lds_w + (size_t)(wave * 16 + row) * WROW;

    // each thread polls exactly one WG's arrival flag (BLOCK == G)
    unsigned* my_flag   = flags + (size_t)wg  * FLAG_STRIDE;
    unsigned* poll_flag = flags + (size_t)tid * FLAG_STRIDE;

    for (int t = 0; t < TSTEPS; ++t) {
        const float* h_rd = h_glob + ((t & 1) ? H : 0);
        float*       h_wr = h_glob + ((t & 1) ? 0 : H);

        // previous h -> LDS fp16: each thread owns 16 contiguous elements
        // (4x global_load_b128, one 32B LDS store) to minimize post-barrier
        // critical-path issue count.
        {
            const float4* src = (const float4*)h_rd + (size_t)tid * 4;
            const float4 f0 = src[0], f1 = src[1], f2 = src[2], f3 = src[3];
            v16h hv;
            hv[0]  = (_Float16)f0.x; hv[1]  = (_Float16)f0.y;
            hv[2]  = (_Float16)f0.z; hv[3]  = (_Float16)f0.w;
            hv[4]  = (_Float16)f1.x; hv[5]  = (_Float16)f1.y;
            hv[6]  = (_Float16)f1.z; hv[7]  = (_Float16)f1.w;
            hv[8]  = (_Float16)f2.x; hv[9]  = (_Float16)f2.y;
            hv[10] = (_Float16)f2.z; hv[11] = (_Float16)f2.w;
            hv[12] = (_Float16)f3.x; hv[13] = (_Float16)f3.y;
            hv[14] = (_Float16)f3.z; hv[15] = (_Float16)f3.w;
            *(v16h*)(lds_h + (size_t)tid * 16) = hv;
        }
        __syncthreads();

        // 16 gate-row dot products per wave via WMMA, broadcast-B columns
        v8f acc = {0.f, 0.f, 0.f, 0.f, 0.f, 0.f, 0.f, 0.f};
        for (int kc = 0; kc < H; kc += 32) {
            v8h a0 = *(const v8h*)(wrow + kc +     hi * 16);
            v8h a1 = *(const v8h*)(wrow + kc + 8 + hi * 16);
            v16h A = __builtin_shufflevector(a0, a1,
                        0,1,2,3,4,5,6,7,8,9,10,11,12,13,14,15);
            // B 32x16: every column holds the same h chunk; lanes 0-15 carry
            // K=kc..kc+15, lanes 16-31 carry K=kc+16..kc+31 (lane-uniform ->
            // LDS broadcast read).
            v16h B = *(const v16h*)(lds_h + kc + hi * 16);
            acc = __builtin_amdgcn_wmma_f32_16x16x32_f16(
                      false, A, false, B, (short)0, acc, false, false);
        }
        // D[m][n] identical across n; take column n==0 (lanes 0 and 16):
        // VGPR v of lane0 holds rows 0..7, of lane16 holds rows 8..15.
        if ((lane & 15) == 0) {
            #pragma unroll
            for (int v = 0; v < 8; ++v)
                lds_gate[wave * 16 + hi * 8 + v] = acc[v];
        }
        __syncthreads();

        // elementwise LSTM cell for this WG's 16 h-elements
        if (tid < JPW) {
            const bool edge = (t == 0) || (t == TSTEPS - 1);
            const float x0 = edge ? 0.0f : seq[(t - 1) * 3 + 0];
            const float x1 = edge ? 0.0f : seq[(t - 1) * 3 + 1];
            const float x2 = edge ? 1.0f : seq[(t - 1) * 3 + 2];
            float gv[4];
            #pragma unroll
            for (int q = 0; q < 4; ++q) {
                const int lr = q * 16 + tid;
                gv[q] = lds_gate[lr] + lds_bias[lr]
                      + lds_wih[lr*3 + 0] * x0
                      + lds_wih[lr*3 + 1] * x1
                      + lds_wih[lr*3 + 2] * x2;
            }
            const float ig = sigmoidf_(gv[0]);
            const float fg = sigmoidf_(gv[1]);
            const float gg = tanhf(gv[2]);
            const float og = sigmoidf_(gv[3]);
            const float c  = fg * lds_c[tid] + ig * gg;
            lds_c[tid] = c;
            const float h = og * tanhf(c);
            const int gj = wg * JPW + tid;
            h_wr[gj] = h;   // regular store: this is the hot L2 exchange line
            // streamed outputs: non-temporal, never read again by this kernel
            if (t >= 1)
                __builtin_nontemporal_store(
                    h, &out[(size_t)H + (size_t)(t - 1) * H + gj]);
            if (t == TSTEPS - 1)
                __builtin_nontemporal_store(h, &out[gj]);
            __threadfence();   // release h_wr device-wide before arrival store
        }
        __syncthreads();

        // ---- distributed grid barrier: one padded release-store per WG,
        //      128 flags polled fully in parallel (one per thread)
        if (tid == 0)
            __hip_atomic_store(my_flag, (unsigned)(t + 1),
                               __ATOMIC_RELEASE, __HIP_MEMORY_SCOPE_AGENT);
        while (__hip_atomic_load(poll_flag, __ATOMIC_RELAXED,
                                 __HIP_MEMORY_SCOPE_AGENT) < (unsigned)(t + 1)) {
        }
        __syncthreads();
        __threadfence();       // acquire other WGs' h writes
    }
}

extern "C" void kernel_launch(void* const* d_in, const int* in_sizes, int n_in,
                              void* d_out, int out_size, void* d_ws, size_t ws_size,
                              hipStream_t stream) {
    const float* seq  = (const float*)d_in[0];
    const float* W_ih = (const float*)d_in[1];
    const float* W_hh = (const float*)d_in[2];
    const float* b_ih = (const float*)d_in[3];
    const float* b_hh = (const float*)d_in[4];
    float* out = (float*)d_out;

    unsigned* flags  = (unsigned*)d_ws;                       // G*FLAG_STRIDE u32 (16 KB)
    float*    h_glob = (float*)((char*)d_ws + (size_t)G * FLAG_STRIDE * 4);

    const size_t smem_bytes = (size_t)ROWS * WROW * 2   // weights fp16
                            + (size_t)H * 2             // h fp16
                            + (64 + 16 + 64 + 192) * 4; // gates, c, bias, wih
    (void)hipFuncSetAttribute((const void*)lstm_persistent,
                              hipFuncAttributeMaxDynamicSharedMemorySize,
                              (int)smem_bytes);

    lstm_init<<<(G * FLAG_STRIDE + 255) / 256, 256, 0, stream>>>(h_glob, flags);
    lstm_persistent<<<G, BLOCK, smem_bytes, stream>>>(
        seq, W_ih, W_hh, b_ih, b_hh, out, h_glob, flags);
}
